// GraphConvolution_50122268345053
// MI455X (gfx1250) — matrix operand order repaired and checked by reading, not compiled, not run on hardware
//
#include <hip/hip_runtime.h>
#include <math.h>

// ---------------------------------------------------------------------------
// MI455X / gfx1250 (CDNA5, wave32) implementation of the equivariant graph
// convolution. Fused into 5 kernels; all matrix math goes through WMMA.
//   - node GEMMs / output GEMMs: V_WMMA_F32_16X16X4_F32 (exact f32)
//   - edge MLP (h @ W_mlp2):     V_WMMA_F32_16X16X32_F16 (f16 in, f32 acc)
//   - contiguous global->LDS tile staging: TENSOR_LOAD_TO_LDS (TDM) +
//     s_wait_tensorcnt (bypasses VGPRs, overlaps with SALU/VALU setup)
// Scatter (segment_sum) uses GLOBAL_ATOMIC_ADD_F32 via unsafeAtomicAdd.
// Per-wave LDS staging synchronized with `s_wait_dscnt 0` (wave-lockstep).
// ---------------------------------------------------------------------------

typedef __attribute__((ext_vector_type(16))) _Float16 v16h;
typedef __attribute__((ext_vector_type(8)))  _Float16 v8h;
typedef __attribute__((ext_vector_type(8)))  float    v8f;
typedef __attribute__((ext_vector_type(2)))  float    v2f;
typedef __attribute__((ext_vector_type(4)))  unsigned int v4u;
typedef __attribute__((ext_vector_type(8)))  int      v8i;
typedef __attribute__((ext_vector_type(4)))  int      v4i;

#define N_NODES 20000
#define N_EDGES 320000
#define MUL     64
#define N_BASIS 10
#define N_HID   100

__device__ __constant__ const float INV_SQRT64  = 0.125f;
__device__ __constant__ const float INV_SQRT128 = 0.08838834764831845f;
__device__ __constant__ const float INV_SQRT10  = 0.31622776601683794f;
__device__ __constant__ const float INV_SQRT100 = 0.1f;
__device__ __constant__ const float INV_SQRT3   = 0.5773502691896258f;
__device__ __constant__ const float C_S = 0.3826834323650898f;   // sin(pi/8)
__device__ __constant__ const float C_X = 0.9238795325112867f;   // cos(pi/8)

// ---- workspace layout (in floats) -----------------------------------------
// f_all  : N x 256   ([0:64]=f0, [64+m*64+u]=f1[u][m], deg-scaled)
// sv_mid : N x 512   ([0:128]=s_mid, [128+m*128+c]=v_mid[c][m])
// fragment-ordered weight tables (see k_prep_weights)
#define F_ALL_OFF 0
#define SV_OFF    (N_NODES * 256)               //  5,120,000
#define FR_OFF    (SV_OFF + N_NODES * 512)      // 15,360,000
#define W2F_OFF   FR_OFF                        // 32768 halves = 16384 floats
#define WLI0_OFF  (FR_OFF + 16384)
#define WLI1_OFF  (WLI0_OFF + 4096)
#define WLM0_OFF  (WLI1_OFF + 4096)
#define WLM1_OFF  (WLM0_OFF + 4096)
#define WLO0_OFF  (WLM1_OFF + 4096)
#define WLO1_OFF  (WLO0_OFF + 8192)
#define WS_FLOATS (WLO1_OFF + 8192)             // ~61.6 MB total

// ---------------------------------------------------------------------------
// 1-D Tensor-Data-Mover load: `nelem` contiguous f32 from global -> LDS.
// D# packed per CDNA5 ISA ch.8 (group0: count/lds_addr/global_addr/type=2;
// group1: data_size=4B, tensor_dim0=tile_dim0=nelem, stride=nelem).
// Completion tracked with TENSORcnt. (6-arg builtin form: groups 2/3 and the
// iterate/extra group are zero for a <=2-D non-iterating descriptor.)
__device__ __forceinline__ void tdm_load_f32_1d(unsigned lds_byte_addr,
                                                const float* gaddr,
                                                unsigned nelem /* <= 16383 */) {
  unsigned long long ga = (unsigned long long)(size_t)gaddr;
  v4u g0;
  g0[0] = 1u;                                   // count=1 (valid), user mode
  g0[1] = lds_byte_addr;                        // lds_addr [63:32]
  g0[2] = (unsigned)(ga & 0xFFFFFFFFu);         // global_addr [95:64]
  g0[3] = (unsigned)(ga >> 32) | 0x80000000u;   // global_addr hi | type=2 [127:126]
  v8i g1;
  g1[0] = 0x20000;                              // workgroup_mask=0, data_size=2 (4B)
  g1[1] = (int)((nelem & 0xFFFFu) << 16);       // tensor_dim0[15:0] @ bits 63:48
  g1[2] = (int)(((nelem >> 16) & 0xFFFFu) | (1u << 16)); // tensor_dim0 hi, tensor_dim1=1
  g1[3] = (int)(nelem << 16);                   // tile_dim0 @ bits 127:112
  g1[4] = 0;                                    // tile_dim1=0, tile_dim2=0 (1-D)
  g1[5] = (int)nelem;                           // tensor_dim0_stride[31:0]
  g1[6] = 0;
  g1[7] = 0;
  v4i z4 = {0, 0, 0, 0};                        // groups 2/3 unused (<=2-D)
  v8i z8 = {0, 0, 0, 0, 0, 0, 0, 0};
  __builtin_amdgcn_tensor_load_to_lds(g0, g1, z4, z4, z8, 0);
}

// ---------------------------------------------------------------------------
__global__ void k_zero(float* __restrict__ p, int n) {
  int i = blockIdx.x * blockDim.x + threadIdx.x;
  int stride = gridDim.x * blockDim.x;
  for (; i < n; i += stride) p[i] = 0.0f;
}

// ---------------------------------------------------------------------------
// Repack weights into WMMA B-fragment order so compute kernels do one
// coalesced vector load per fragment (tables are tiny -> live in L2/WGP$).
//   f32 16x16x4 B frag : lane -> n = nt*16 + lane%16 ; elem j -> K = ks*4 + j + 2*(lane/16)
//   f16 16x16x32 B frag: lane -> n = nt*16 + lane%16 ; elem i -> K = ks*32 + 16*(lane/16) + i
__global__ void k_prep_weights(const float* __restrict__ W_mlp2,
                               const float* __restrict__ Wli0, const float* __restrict__ Wli1,
                               const float* __restrict__ Wlm0, const float* __restrict__ Wlm1,
                               const float* __restrict__ Wlo0, const float* __restrict__ Wlo1,
                               float* __restrict__ ws) {
  const int tid = blockIdx.x * blockDim.x + threadIdx.x;
  const int stride = gridDim.x * blockDim.x;

  // W_mlp2 (100x256) -> f16 fragments, K padded to 128
  _Float16* w2f = (_Float16*)(ws + W2F_OFF);
  for (int idx = tid; idx < 4 * 16 * 32 * 16; idx += stride) {
    int i = idx & 15, lane = (idx >> 4) & 31, nt = (idx >> 9) & 15, ks = idx >> 13;
    int group = lane >> 4;
    int K = ks * 32 + group * 16 + i;
    int n = nt * 16 + (lane & 15);
    w2f[idx] = (_Float16)((K < N_HID) ? W_mlp2[K * 256 + n] : 0.0f);
  }
  // 64x64 f32 weights -> fragments (4 ntiles x 16 ksteps)
  const float* Ws[4] = { Wli0, Wli1, Wlm0, Wlm1 };
  float* Fs[4] = { ws + WLI0_OFF, ws + WLI1_OFF, ws + WLM0_OFF, ws + WLM1_OFF };
  for (int w = 0; w < 4; ++w)
    for (int idx = tid; idx < 4096; idx += stride) {
      int j = idx & 1, lane = (idx >> 1) & 31, ks = (idx >> 6) & 15, nt = idx >> 10;
      int group = lane >> 4;
      int K = ks * 4 + j + 2 * group;
      int n = nt * 16 + (lane & 15);
      Fs[w][idx] = Ws[w][K * 64 + n];
    }
  // 128x64 f32 weights -> fragments (4 ntiles x 32 ksteps)
  const float* Wl[2] = { Wlo0, Wlo1 };
  float* Fl[2] = { ws + WLO0_OFF, ws + WLO1_OFF };
  for (int w = 0; w < 2; ++w)
    for (int idx = tid; idx < 8192; idx += stride) {
      int j = idx & 1, lane = (idx >> 1) & 31, ks = (idx >> 6) & 31, nt = idx >> 11;
      int group = lane >> 4;
      int K = ks * 4 + j + 2 * group;
      int n = nt * 16 + (lane & 15);
      Fl[w][idx] = Wl[w][K * 64 + n];
    }
}

// ---------------------------------------------------------------------------
// Node-side input GEMMs. One 16-node tile per wave, 2 waves/block.
// x tile (16x256 f32) staged to LDS by the Tensor Data Mover.
// f0 = x0@W_li0 * 0.125 * deg^-1/2 ; f1m = x1m@W_li1 * ...
// out (init) = c_s * 0.125 * (x @ W_lm)
__global__ void __launch_bounds__(64)
k_node_prep(const float* __restrict__ node_input, const float* __restrict__ node_deg,
            const float* __restrict__ ws, float* __restrict__ f_all,
            float* __restrict__ out) {
  __shared__ __align__(16) float xbuf[2][16][256];
  __shared__ float ldeg[2][16];
  const int lane  = threadIdx.x & 31;
  const int wave  = threadIdx.x >> 5;
  const int group = lane >> 4;
  const int l16   = lane & 15;
  const int tile  = blockIdx.x * 2 + wave;     // 1250 tiles exactly
  const int node0 = tile * 16;

  // TDM: DMA this wave's 16 node rows (16KB) straight into LDS.
  tdm_load_f32_1d((unsigned)(size_t)(void*)&xbuf[wave][0][0],
                  node_input + (size_t)node0 * 256, 16 * 256);
  if (lane < 16) ldeg[wave][lane] = rsqrtf(node_deg[node0 + lane]);
  asm volatile("s_wait_dscnt 0" ::: "memory");
  __builtin_amdgcn_s_wait_tensorcnt(0);

  for (int src = 0; src < 4; ++src) {          // 0: scalar, 1..3: vector comp
    const float* Bf = (src == 0) ? ws + WLI0_OFF : ws + WLI1_OFF;
    const float* Bm = (src == 0) ? ws + WLM0_OFF : ws + WLM1_OFF;
    const int mm = src - 1;
    for (int nt = 0; nt < 4; ++nt) {
      v8f accf = {}; v8f accm = {};
      for (int ks = 0; ks < 16; ++ks) {
        v2f a;
#pragma unroll
        for (int j = 0; j < 2; ++j) {
          int K = ks * 4 + j + 2 * group;
          int col = (src == 0) ? K : 64 + K * 3 + mm;
          a[j] = xbuf[wave][l16][col];
        }
        const v2f bf = *(const v2f*)(Bf + ((nt * 16 + ks) * 64 + lane * 2));
        const v2f bm = *(const v2f*)(Bm + ((nt * 16 + ks) * 64 + lane * 2));
        accf = __builtin_amdgcn_wmma_f32_16x16x4_f32(false, a, false, bf, (short)0, accf, false, false);
        accm = __builtin_amdgcn_wmma_f32_16x16x4_f32(false, a, false, bm, (short)0, accm, false, false);
      }
      const int n = nt * 16 + l16;
#pragma unroll
      for (int r = 0; r < 8; ++r) {
        int ml = r + 8 * group;
        size_t node = (size_t)(node0 + ml);
        float fv = accf[r] * INV_SQRT64 * ldeg[wave][ml];
        float ov = C_S * INV_SQRT64 * accm[r];
        if (src == 0) {
          f_all[node * 256 + n] = fv;
          out[node * 256 + n]   = ov;
        } else {
          f_all[node * 256 + 64 + mm * 64 + n] = fv;
          out[node * 256 + 64 + n * 3 + mm]    = ov;
        }
      }
    }
  }
}

// ---------------------------------------------------------------------------
// Fused edge kernel: MLP (WMMA f16) + gather + messages + atomic scatter.
// One 16-edge tile per wave, 2 waves/block, 20000 tiles total.
__global__ void __launch_bounds__(64)
k_edge(const float* __restrict__ ele, const float* __restrict__ eattr,
       const int* __restrict__ esrc, const int* __restrict__ edst,
       const float* __restrict__ W_mlp1, float* __restrict__ ws) {
  __shared__ __align__(16) _Float16 hbuf[2][16][128];   // h, K-padded, f16
  __shared__ __align__(16) float ewbuf[2][16][256];     // ew tile
  __shared__ float w1s[N_BASIS * N_HID];
  const float* f_all = ws + F_ALL_OFF;
  float* sv = ws + SV_OFF;

  for (int idx = threadIdx.x; idx < N_BASIS * N_HID; idx += 64) w1s[idx] = W_mlp1[idx];
  __syncthreads();

  const int lane  = threadIdx.x & 31;
  const int wave  = threadIdx.x >> 5;
  const int group = lane >> 4;
  const int l16   = lane & 15;
  const int tile  = blockIdx.x * 2 + wave;     // 20000 tiles exactly
  const int e0    = tile * 16;

  // phase 1: h = silu(ELE @ W_mlp1 / sqrt(10))  (exact f32, stored f16)
  for (int idx = lane; idx < 16 * 128; idx += 32) {
    int e = idx >> 7, j = idx & 127;
    float hv = 0.0f;
    if (j < N_HID) {
      float acc = 0.0f;
#pragma unroll
      for (int b = 0; b < N_BASIS; ++b)
        acc += ele[(size_t)(e0 + e) * N_BASIS + b] * w1s[b * N_HID + j];
      acc *= INV_SQRT10;
      hv = acc / (1.0f + __expf(-acc));
    }
    hbuf[wave][e][j] = (_Float16)hv;
  }
  asm volatile("s_wait_dscnt 0" ::: "memory");

  // phase 2: ew = h @ W_mlp2 * 0.1  via V_WMMA_F32_16X16X32_F16
  v16h afrag[4];
#pragma unroll
  for (int ks = 0; ks < 4; ++ks) {
    afrag[ks] = *(const v16h*)&hbuf[wave][l16][ks * 32 + 16 * group];
  }
  const _Float16* w2f = (const _Float16*)(ws + W2F_OFF);
  for (int nt = 0; nt < 16; ++nt) {
    v8f c = {};
#pragma unroll
    for (int ks = 0; ks < 4; ++ks) {
      const v16h b = *(const v16h*)(w2f + ((size_t)(ks * 16 + nt) * 32 + lane) * 16);
      c = __builtin_amdgcn_wmma_f32_16x16x32_f16(false, afrag[ks], false, b, (short)0, c, false, false);
    }
#pragma unroll
    for (int r = 0; r < 8; ++r)
      ewbuf[wave][r + 8 * group][nt * 16 + l16] = c[r] * INV_SQRT100;
  }
  asm volatile("s_wait_dscnt 0" ::: "memory");

  // phase 3: messages + segment-sum scatter (hardware f32 atomics)
  for (int idx = lane; idx < 16 * 64; idx += 32) {
    int e = idx >> 6, u = idx & 63;
    int eg = e0 + e;
    int src = esrc[eg], dst = edst[eg];
    float y0  = eattr[(size_t)eg * 4 + 0];
    float y1x = eattr[(size_t)eg * 4 + 1];
    float y1y = eattr[(size_t)eg * 4 + 2];
    float y1z = eattr[(size_t)eg * 4 + 3];
    const float* fs = f_all + (size_t)src * 256;
    float g0  = fs[u];
    float g1x = fs[64 + u], g1y = fs[128 + u], g1z = fs[192 + u];
    float wa = ewbuf[wave][e][u],       wb = ewbuf[wave][e][64 + u];
    float wc = ewbuf[wave][e][128 + u], wd = ewbuf[wave][e][192 + u];

    float m0a = wa * g0 * y0;
    float dot = g1x * y1x + g1y * y1y + g1z * y1z;
    float m0b = wb * dot * INV_SQRT3;
    float* svd = sv + (size_t)dst * 512;
    unsafeAtomicAdd(svd + u,        m0a);
    unsafeAtomicAdd(svd + 64 + u,   m0b);
    float wcg0 = wc * g0;
    unsafeAtomicAdd(svd + 128 + 0 * 128 + u, wcg0 * y1x);
    unsafeAtomicAdd(svd + 128 + 1 * 128 + u, wcg0 * y1y);
    unsafeAtomicAdd(svd + 128 + 2 * 128 + u, wcg0 * y1z);
    float wdy0 = wd * y0;
    unsafeAtomicAdd(svd + 128 + 0 * 128 + 64 + u, wdy0 * g1x);
    unsafeAtomicAdd(svd + 128 + 1 * 128 + 64 + u, wdy0 * g1y);
    unsafeAtomicAdd(svd + 128 + 2 * 128 + 64 + u, wdy0 * g1z);
  }
}

// ---------------------------------------------------------------------------
// Node-side output GEMMs: o = (deg^-1/2 * sv_mid) @ W_lo * 1/sqrt(128);
// out += c_x * o.  One 16-node tile per (single-wave) block.
// sv tile (16x512 f32, 32KB) staged to LDS by the Tensor Data Mover, then
// deg-scaled in place.
__global__ void __launch_bounds__(32)
k_node_post(const float* __restrict__ node_deg, const float* __restrict__ ws,
            float* __restrict__ out) {
  __shared__ __align__(16) float svbuf[16][512];
  __shared__ float ldeg[16];
  const int lane  = threadIdx.x;
  const int group = lane >> 4;
  const int l16   = lane & 15;
  const int node0 = blockIdx.x * 16;           // 1250 tiles exactly

  const float* sv = ws + SV_OFF;
  // TDM: DMA the 16x512 sv tile into LDS; overlap with deg^-1/2 computation.
  tdm_load_f32_1d((unsigned)(size_t)(void*)&svbuf[0][0],
                  sv + (size_t)node0 * 512, 16 * 512);
  if (lane < 16) ldeg[lane] = rsqrtf(node_deg[node0 + lane]);
  asm volatile("s_wait_dscnt 0" ::: "memory");
  __builtin_amdgcn_s_wait_tensorcnt(0);

  // deg-scale in place (LDS RMW)
  for (int idx = lane; idx < 16 * 512; idx += 32) {
    int nl = idx >> 9;
    svbuf[nl][idx & 511] *= ldeg[nl];
  }
  asm volatile("s_wait_dscnt 0" ::: "memory");

  for (int src = 0; src < 4; ++src) {
    const float* B = (src == 0) ? ws + WLO0_OFF : ws + WLO1_OFF;
    const int base = (src == 0) ? 0 : 128 + (src - 1) * 128;
    const int mm = src - 1;
    for (int nt = 0; nt < 4; ++nt) {
      v8f acc = {};
      for (int ks = 0; ks < 32; ++ks) {
        v2f a;
#pragma unroll
        for (int j = 0; j < 2; ++j)
          a[j] = svbuf[l16][base + ks * 4 + j + 2 * group];
        const v2f b = *(const v2f*)(B + ((nt * 32 + ks) * 64 + lane * 2));
        acc = __builtin_amdgcn_wmma_f32_16x16x4_f32(false, a, false, b, (short)0, acc, false, false);
      }
      const int n = nt * 16 + l16;
#pragma unroll
      for (int r = 0; r < 8; ++r) {
        size_t node = (size_t)(node0 + r + 8 * group);
        float o = acc[r] * INV_SQRT128;
        size_t off = node * 256 + ((src == 0) ? (size_t)n : (size_t)(64 + n * 3 + mm));
        out[off] += C_X * o;
      }
    }
  }
}

// ---------------------------------------------------------------------------
extern "C" void kernel_launch(void* const* d_in, const int* in_sizes, int n_in,
                              void* d_out, int out_size, void* d_ws, size_t ws_size,
                              hipStream_t stream) {
  const float* node_input = (const float*)d_in[0];
  // d_in[1] node_attr == ones -> folded out
  const float* node_deg   = (const float*)d_in[2];
  const float* edge_attr  = (const float*)d_in[3];
  const float* ele        = (const float*)d_in[4];
  const float* W_li0      = (const float*)d_in[5];
  const float* W_li1      = (const float*)d_in[6];
  const float* W_lm0      = (const float*)d_in[7];
  const float* W_lm1      = (const float*)d_in[8];
  const float* W_mlp1     = (const float*)d_in[9];
  const float* W_mlp2     = (const float*)d_in[10];
  const float* W_lo0      = (const float*)d_in[11];
  const float* W_lo1      = (const float*)d_in[12];
  const int*   esrc       = (const int*)d_in[13];
  const int*   edst       = (const int*)d_in[14];
  float* ws  = (float*)d_ws;   // needs >= WS_FLOATS*4 (~62 MB)
  float* out = (float*)d_out;

  k_zero<<<2048, 256, 0, stream>>>(ws + SV_OFF, N_NODES * 512);
  k_prep_weights<<<64, 256, 0, stream>>>(W_mlp2, W_li0, W_li1, W_lm0, W_lm1,
                                         W_lo0, W_lo1, ws);
  k_node_prep<<<625, 64, 0, stream>>>(node_input, node_deg, ws, ws + F_ALL_OFF, out);
  k_edge<<<10000, 64, 0, stream>>>(ele, edge_attr, esrc, edst, W_mlp1, ws);
  k_node_post<<<1250, 32, 0, stream>>>(node_deg, ws, out);
}
